// LanguageModel_25374666785579
// MI455X (gfx1250) — matrix operand order repaired and checked by reading, not compiled, not run on hardware
//
#include <hip/hip_runtime.h>
#include <hip/hip_bf16.h>

typedef unsigned short u16;
typedef __attribute__((ext_vector_type(16))) __bf16 v16bf;
typedef __attribute__((ext_vector_type(8)))  __bf16 v8bf;
typedef __attribute__((ext_vector_type(8)))  float  v8f;

// ---- constants ----
#define Bsz   64
#define Tlen  512
#define T2    514
#define Hdim  256
#define Vocab 5000
#define VocabPad 5056      // 79 * 64, zero-padded rows in workspace copy of W_out
#define KSTEPS 8           // 256 / 32

// round-to-nearest-even f32 -> bf16
__device__ __forceinline__ u16 f2bf(float f) {
  unsigned u = __float_as_uint(f);
  u += 0x7FFFu + ((u >> 16) & 1u);
  return (u16)(u >> 16);
}

// A fragment (16x32 bf16, M rows per lane): lane l (row = l&15, half = l>>4)
// holds K = {8*half .. 8*half+7} in v[0..3] pairs and {16+8*half ..} in v[4..7].
__device__ __forceinline__ v16bf load_a_frag(const u16* rowptr, int half) {
  const v8bf lo = *(const v8bf*)(rowptr + 8 * half);
  const v8bf hi = *(const v8bf*)(rowptr + 16 + 8 * half);
  v16bf a;
#pragma unroll
  for (int i = 0; i < 8; ++i) { a[i] = lo[i]; a[8 + i] = hi[i]; }
  return a;
}

// B fragment (32x16 bf16, col = lane&15): K = {8*half .. 8*half+7} in b[0..7]
// and {16+8*half ..} in b[8..15], read from row-major W[n, k] row pointer (W row n == column n of B).
__device__ __forceinline__ v16bf load_b_frag(const u16* rowptr, int half) {
  const v8bf lo = *(const v8bf*)(rowptr + 8 * half);
  const v8bf hi = *(const v8bf*)(rowptr + 16 + 8 * half);
  v16bf b;
#pragma unroll
  for (int i = 0; i < 8; ++i) { b[i] = lo[i]; b[8 + i] = hi[i]; }
  return b;
}

// -------- f32 -> bf16 conversion with zero padding to n_dst elements --------
__global__ void f32_to_bf16_kernel(const float* __restrict__ src,
                                   u16* __restrict__ dst, int n_src, int n_dst) {
  for (int i = blockIdx.x * blockDim.x + threadIdx.x; i < n_dst;
       i += gridDim.x * blockDim.x)
    dst[i] = (i < n_src) ? f2bf(src[i]) : (u16)0;
}

// -------------------- embedding gather -> bf16 --------------------
__global__ void embed_kernel(const int* __restrict__ x,
                             const float* __restrict__ emb,
                             u16* __restrict__ X0) {
  long long n = (long long)blockIdx.x * blockDim.x + threadIdx.x;
  if (n >= (long long)Bsz * Tlen * Hdim) return;
  int e = (int)(n & (Hdim - 1));
  long long pos = n >> 8;            // Hdim == 256
  int t = (int)(pos & (Tlen - 1));
  int b = (int)(pos >> 9);           // Tlen == 512
  int tok = x[b * T2 + t];
  X0[n] = f2bf(emb[(long long)tok * Hdim + e]);
}

// -------------------- generic bf16 GEMM: C = A * W^T + bias --------------------
// A: M x 256 (bf16 row-major), W: >= nGroups*64 x 256 (bf16 row-major,
// zero-padded past N), C: M x N (f32).  One wave computes a 16(M) x 64(N)
// tile; K = 256 in 8 unconditional WMMA steps (no branches in the K-loop).
__global__ __launch_bounds__(256) void gemm_bias_kernel(
    const u16* __restrict__ A, const u16* __restrict__ W,
    const float* __restrict__ bias, float* __restrict__ C,
    int M, int N, int nGroups) {
  int gw   = (blockIdx.x * blockDim.x + threadIdx.x) >> 5;
  int lane = threadIdx.x & 31;
  int mtile = gw / nGroups;
  int ngrp  = gw % nGroups;
  if (mtile * 16 >= M) return;
  const int half = lane >> 4, l16 = lane & 15;
  const int arow = mtile * 16 + l16;
  const int nrow0 = ngrp * 64 + l16;

  v8f acc[4] = {};
#pragma unroll
  for (int ks = 0; ks < KSTEPS; ++ks) {
    v16bf a = load_a_frag(A + arow * Hdim + ks * 32, half);
#pragma unroll
    for (int j = 0; j < 4; ++j) {
      v16bf b = load_b_frag(W + (nrow0 + j * 16) * Hdim + ks * 32, half);
      acc[j] = __builtin_amdgcn_wmma_f32_16x16x32_bf16(
          false, a, false, b, (short)0, acc[j], false, false);
    }
  }
#pragma unroll
  for (int j = 0; j < 4; ++j) {
    int n = nrow0 + j * 16;
    if (n >= N) continue;            // ragged edge only at store time
    float bv = bias[n];
#pragma unroll
    for (int r = 0; r < 8; ++r) {
      int m = mtile * 16 + r + 8 * half;
      C[(long long)m * N + n] = acc[j][r] + bv;
    }
  }
}

// -------------------- persistent recurrence --------------------
// h_{t} = tanh(pre_t + h_{t-1} @ Whh^T + bhh).  Recurrence is independent
// across batch rows, so block bi owns batch rows [16*bi, 16*bi+16).
// Whh (256x256 bf16) lives in LDS with padded stride 264 (bank-conflict free);
// h lives in LDS (16x264 bf16).  8 waves x 2 h-tiles x 8 k-steps = 128 WMMA/step.
#define LDP 264
__global__ __launch_bounds__(256) void rnn_layer_kernel(
    const float* __restrict__ PRE,   // [B*T, H] f32
    const u16*   __restrict__ WhhB,  // [H, H]  bf16
    const float* __restrict__ bhh,   // [H]
    u16*         __restrict__ Xout,  // [B*T, H] bf16
    float*       __restrict__ hlast) // [B, H]  f32 (this layer's slot)
{
  __shared__ u16 whh_s[Hdim * LDP];
  __shared__ u16 hold[16 * LDP];
  const int tid = threadIdx.x;
  const int b0  = blockIdx.x * 16;

  // stage Whh into LDS (8-byte chunks), padded rows
  for (int i = tid; i < Hdim * 64; i += 256) {
    int row = i >> 6, c = (i & 63) * 4;
    *(uint2*)(&whh_s[row * LDP + c]) = *(const uint2*)(WhhB + row * Hdim + c);
  }
  for (int i = tid; i < 16 * LDP; i += 256) hold[i] = 0;   // h0 = 0
  __syncthreads();

  const int lane = tid & 31, w = tid >> 5;
  const int half = lane >> 4, l16 = lane & 15;
  const int hT0  = w * 2;                 // 2 output h-tiles per wave

  for (int t = 0; t < Tlen; ++t) {
    v8f acc[2] = {};
#pragma unroll
    for (int ks = 0; ks < KSTEPS; ++ks) {
      v16bf a = load_a_frag(&hold[l16 * LDP + ks * 32], half);
#pragma unroll
      for (int j = 0; j < 2; ++j) {
        int nrow = (hT0 + j) * 16 + l16;
        v16bf b = load_b_frag(&whh_s[nrow * LDP + ks * 32], half);
        acc[j] = __builtin_amdgcn_wmma_f32_16x16x32_bf16(
            false, a, false, b, (short)0, acc[j], false, false);
      }
    }
    float hv[2][8];
#pragma unroll
    for (int j = 0; j < 2; ++j) {
      int n = (hT0 + j) * 16 + l16;
      float bb = bhh[n];
#pragma unroll
      for (int r = 0; r < 8; ++r) {
        int m = r + 8 * half;                              // local batch row
        long long gi = ((long long)(b0 + m) * Tlen + t) * Hdim + n;
        float v = tanhf(acc[j][r] + PRE[gi] + bb);
        hv[j][r] = v;
        Xout[gi] = f2bf(v);
        if (t == Tlen - 1) hlast[(long long)(b0 + m) * Hdim + n] = v;
      }
    }
    __syncthreads();                        // all reads of hold complete
#pragma unroll
    for (int j = 0; j < 2; ++j) {
      int n = (hT0 + j) * 16 + l16;
#pragma unroll
      for (int r = 0; r < 8; ++r)
        hold[(r + 8 * half) * LDP + n] = f2bf(hv[j][r]);
    }
    __syncthreads();                        // h_t visible to everyone
  }
}

// -------------------- seq_len tail --------------------
__global__ void tail_kernel(const int* __restrict__ x, float* __restrict__ seq) {
  int b = threadIdx.x;
  if (b < Bsz) seq[b] = (float)x[b * T2 + (T2 - 1)];
}

extern "C" void kernel_launch(void* const* d_in, const int* in_sizes, int n_in,
                              void* d_out, int out_size, void* d_ws, size_t ws_size,
                              hipStream_t stream) {
  (void)in_sizes; (void)n_in; (void)out_size; (void)ws_size;
  const int*   x     = (const int*)d_in[0];
  const float* emb   = (const float*)d_in[1];
  const float* W_ih  = (const float*)d_in[2];   // [2,256,256]
  const float* W_hh  = (const float*)d_in[3];   // [2,256,256]
  const float* b_ih  = (const float*)d_in[4];   // [2,256]
  const float* b_hh  = (const float*)d_in[5];   // [2,256]
  const float* W_out = (const float*)d_in[6];   // [5000,256]
  const float* b_out = (const float*)d_in[7];   // [5000]

  // ---- workspace layout (bytes) ----
  char* ws = (char*)d_ws;
  u16*   WIHb  = (u16*)(ws + 0);                 //   131072 bf16
  u16*   WHHb  = (u16*)(ws + 262144);            //   131072 bf16
  u16*   WOUTb = (u16*)(ws + 524288);            //  5056*256 bf16 (zero-padded)
  u16*   X0    = (u16*)(ws + 3145728);           //  8388608 bf16 (emb / layer-1 out)
  u16*   X1    = (u16*)(ws + 20971520);          //  8388608 bf16 (layer-0 out)
  float* PREb  = (float*)(ws + 41943040);        //  8388608 f32

  // ---- output layout ----
  const long long LOGITS = (long long)Bsz * Tlen * Vocab;  // 163,840,000
  float* d_logits = (float*)d_out;
  float* d_last   = d_logits + LOGITS;           // [2,64,256]
  float* d_seq    = d_last + 2 * Bsz * Hdim;     // [64]

  const int M = Bsz * Tlen;                      // 32768

  // 1) weights -> bf16 (W_out zero-padded to 5056 rows)
  f32_to_bf16_kernel<<<512, 256, 0, stream>>>(W_ih, WIHb,
                                              2 * Hdim * Hdim, 2 * Hdim * Hdim);
  f32_to_bf16_kernel<<<512, 256, 0, stream>>>(W_hh, WHHb,
                                              2 * Hdim * Hdim, 2 * Hdim * Hdim);
  f32_to_bf16_kernel<<<5056, 256, 0, stream>>>(W_out, WOUTb,
                                               Vocab * Hdim, VocabPad * Hdim);

  // 2) embedding gather
  embed_kernel<<<(Bsz * Tlen * Hdim) / 256, 256, 0, stream>>>(x, emb, X0);

  // 3) layer 0:  pre = X0 @ W_ih0^T + b_ih0 ;  recurrence
  gemm_bias_kernel<<<1024, 256, 0, stream>>>(X0, WIHb, b_ih, PREb, M, Hdim, 4);
  rnn_layer_kernel<<<4, 256, 0, stream>>>(PREb, WHHb, b_hh, X1, d_last);

  // 4) layer 1
  gemm_bias_kernel<<<1024, 256, 0, stream>>>(X1, WIHb + Hdim * Hdim,
                                             b_ih + Hdim, PREb, M, Hdim, 4);
  rnn_layer_kernel<<<4, 256, 0, stream>>>(PREb, WHHb + Hdim * Hdim,
                                          b_hh + Hdim, X0, d_last + Bsz * Hdim);

  // 5) logits = out1 @ W_out^T + b_out   (79 groups of 64, padded W, guarded store)
  const int nGroups = VocabPad / 64;                        // 79
  const int waves   = (M / 16) * nGroups;                   // 161792
  gemm_bias_kernel<<<(waves * 32 + 255) / 256, 256, 0, stream>>>(
      X0, WOUTb, b_out, d_logits, M, Vocab, nGroups);

  // 6) seq_len passthrough
  tail_kernel<<<1, 64, 0, stream>>>(x, d_seq);
}
